// EfficientDeformableResidualBlock_32186484916725
// MI455X (gfx1250) — matrix operand order repaired and checked by reading, not compiled, not run on hardware
//
#include <hip/hip_runtime.h>
#include <math.h>

// Problem constants (from reference): B=4, C=256, H=W=64, K=3
#define Bn   4
#define Cc   256
#define Hd   64
#define Wd   64
#define HW   4096          // H*W
#define K2c  9
#define KD   2304          // C*K2
#define NPIX 16384         // B*H*W
#define EPSf 1e-5f

typedef __attribute__((ext_vector_type(16))) _Float16 v16h;
typedef __attribute__((ext_vector_type(8)))  _Float16 v8h;
typedef __attribute__((ext_vector_type(8)))  float    v8f;

// ---------------------------------------------------------------------------
// Cast deform-conv weights (O=256, C*K2=2304) fp32 -> fp16 B-matrices.
// Reduction index r = c*9+k matches dw's native (C,3,3) flattening, so the
// f16 B matrix is just a dtype cast of dw, row-major [N=256][K=2304].
// ---------------------------------------------------------------------------
__global__ void pack_w_kernel(const float* __restrict__ dw1,
                              const float* __restrict__ dw2,
                              _Float16* __restrict__ bt1,
                              _Float16* __restrict__ bt2) {
    int i = blockIdx.x * 256 + threadIdx.x;
    if (i < Cc * KD) {
        bt1[i] = (_Float16)dw1[i];
        bt2[i] = (_Float16)dw2[i];
    }
}

__global__ void zero_stats_kernel(float* __restrict__ stats) {
    if (threadIdx.x < 24) stats[threadIdx.x] = 0.0f;
}

// ---------------------------------------------------------------------------
// Offset conv: 3x3, C=256 -> 18 channels, zero padding. One thread per pixel,
// 18 accumulators live in registers; weight indices are wave-uniform so the
// compiler can scalarize those loads.
// ---------------------------------------------------------------------------
__global__ void __launch_bounds__(256)
off_conv_kernel(const float* __restrict__ in, const float* __restrict__ ow,
                const float* __restrict__ ob, float* __restrict__ off) {
    int p   = blockIdx.x * 256 + threadIdx.x;     // 0..16383
    int b   = p >> 12;
    int pix = p & 4095;
    int y   = pix >> 6;
    int x   = pix & 63;

    float acc[18];
#pragma unroll
    for (int o = 0; o < 18; ++o) acc[o] = ob[o];

    const float* xb = in + (size_t)b * Cc * HW;
    for (int c = 0; c < Cc; ++c) {
        float v[9];
#pragma unroll
        for (int ky = 0; ky < 3; ++ky)
#pragma unroll
            for (int kx = 0; kx < 3; ++kx) {
                int yy = y + ky - 1, xx = x + kx - 1;
                bool ok = (yy >= 0) & (yy < Hd) & (xx >= 0) & (xx < Wd);
                v[ky * 3 + kx] = ok ? xb[(size_t)c * HW + yy * Wd + xx] : 0.0f;
            }
        const float* wc = ow + (size_t)c * 9;     // ow[(o*C + c)*9 + i]
#pragma unroll
        for (int o = 0; o < 18; ++o) {
            const float* w = wc + (size_t)o * Cc * 9;
#pragma unroll
            for (int i = 0; i < 9; ++i) acc[o] = fmaf(w[i], v[i], acc[o]);
        }
    }
#pragma unroll
    for (int o = 0; o < 18; ++o) off[((size_t)(b * 18 + o)) * HW + pix] = acc[o];
}

// ---------------------------------------------------------------------------
// Bilinear sample + im2col pack into A [M=16384][K=2304] f16.
// One thread per (pixel, tap): corner indices/weights computed once,
// then a 256-channel gather loop (x is L2-resident: 4 MB per batch).
// Validity is folded into the bilinear weights; indices are clamped.
// ---------------------------------------------------------------------------
__global__ void __launch_bounds__(256)
sample_pack_kernel(const float* __restrict__ in, const float* __restrict__ off,
                   _Float16* __restrict__ A) {
    int t = blockIdx.x * 256 + threadIdx.x;       // 0..147455
    int p = t / 9;
    int k = t - p * 9;
    int b   = p >> 12;
    int pix = p & 4095;
    int y   = pix >> 6;
    int x   = pix & 63;

    float dy = off[((size_t)(b * 18 + 2 * k)) * HW + pix];
    float dx = off[((size_t)(b * 18 + 2 * k + 1)) * HW + pix];
    float py = (float)y + (float)(k / 3 - 1) + dy;
    float px = (float)x + (float)(k % 3 - 1) + dx;

    float y0f = floorf(py), x0f = floorf(px);
    float wy1 = py - y0f, wx1 = px - x0f;
    float wy0 = 1.0f - wy1, wx0 = 1.0f - wx1;
    int y0 = (int)y0f, x0 = (int)x0f;
    int y1 = y0 + 1,   x1 = x0 + 1;

    float vy0 = (y0 >= 0 && y0 < Hd) ? 1.0f : 0.0f;
    float vy1 = (y1 >= 0 && y1 < Hd) ? 1.0f : 0.0f;
    float vx0 = (x0 >= 0 && x0 < Wd) ? 1.0f : 0.0f;
    float vx1 = (x1 >= 0 && x1 < Wd) ? 1.0f : 0.0f;
    int y0c = min(max(y0, 0), Hd - 1), y1c = min(max(y1, 0), Hd - 1);
    int x0c = min(max(x0, 0), Wd - 1), x1c = min(max(x1, 0), Wd - 1);

    float w00 = wy0 * wx0 * vy0 * vx0;
    float w01 = wy0 * wx1 * vy0 * vx1;
    float w10 = wy1 * wx0 * vy1 * vx0;
    float w11 = wy1 * wx1 * vy1 * vx1;
    int i00 = y0c * Wd + x0c, i01 = y0c * Wd + x1c;
    int i10 = y1c * Wd + x0c, i11 = y1c * Wd + x1c;

    const float* xb = in + (size_t)b * Cc * HW;
    _Float16* dst = A + (size_t)p * KD + k;       // stride 9 over channels
    for (int c = 0; c < Cc; ++c) {
        const float* xc = xb + (size_t)c * HW;
        float v = w00 * xc[i00] + w01 * xc[i01] + w10 * xc[i10] + w11 * xc[i11];
        dst[(size_t)c * 9] = (_Float16)v;
    }
}

// ---------------------------------------------------------------------------
// WMMA GEMM: out[M=16384, N=256] = A[M,K=2304] x Bt[N,K]^T + bias, f16 in,
// f32 accumulate via v_wmma_f32_16x16x32_f16.
// Block: 256 threads = 8 waves; block owns 32 M-rows x full 256 N.
// Wave w handles N in [32w, 32w+32): 2x2 grid of 16x16 accum tiles.
// Fragment layout per ISA 7.12.2 (16-bit A 16x32: lane = M%16, K-half by
// lane>=16; B mirrors with N; C/D: VGPR r -> M = r + 8*(lane>=16), N=lane&15).
// ---------------------------------------------------------------------------
__global__ void __launch_bounds__(256)
gemm_wmma_kernel(const _Float16* __restrict__ A, const _Float16* __restrict__ Bt,
                 const float* __restrict__ bias, float* __restrict__ out) {
    const int lid   = threadIdx.x & 31;
    const int wave  = threadIdx.x >> 5;           // 0..7
    const int mtile = blockIdx.x * 32;            // 512 blocks
    const int nbase = wave * 32;
    const int lm    = lid & 15;
    const int hi    = lid >> 4;                   // lane half selects K group

    const _Float16* a0p = A  + (size_t)(mtile + lm)      * KD + hi * 8;
    const _Float16* a1p = A  + (size_t)(mtile + 16 + lm) * KD + hi * 8;
    const _Float16* b0p = Bt + (size_t)(nbase + lm)      * KD + hi * 8;
    const _Float16* b1p = Bt + (size_t)(nbase + 16 + lm) * KD + hi * 8;

    v8f c00 = {}, c01 = {}, c10 = {}, c11 = {};

    for (int kb = 0; kb < KD; kb += 32) {
        union { v16h v; v8h h[2]; } a0, a1, b0, b1;
        a0.h[0] = *(const v8h*)(a0p);      a0.h[1] = *(const v8h*)(a0p + 16);
        a1.h[0] = *(const v8h*)(a1p);      a1.h[1] = *(const v8h*)(a1p + 16);
        b0.h[0] = *(const v8h*)(b0p);      b0.h[1] = *(const v8h*)(b0p + 16);
        b1.h[0] = *(const v8h*)(b1p);      b1.h[1] = *(const v8h*)(b1p + 16);
        a0p += 32; a1p += 32; b0p += 32; b1p += 32;
        __builtin_prefetch(a0p + 64, 0, 1);       // global_prefetch_b8
        __builtin_prefetch(b0p + 64, 0, 1);

        c00 = __builtin_amdgcn_wmma_f32_16x16x32_f16(false, a0.v, false, b0.v,
                                                     (short)0, c00, false, false);
        c01 = __builtin_amdgcn_wmma_f32_16x16x32_f16(false, a0.v, false, b1.v,
                                                     (short)0, c01, false, false);
        c10 = __builtin_amdgcn_wmma_f32_16x16x32_f16(false, a1.v, false, b0.v,
                                                     (short)0, c10, false, false);
        c11 = __builtin_amdgcn_wmma_f32_16x16x32_f16(false, a1.v, false, b1.v,
                                                     (short)0, c11, false, false);
    }

    // Epilogue: bias + scatter into NCHW h[b][oc][pix].
    int n0 = nbase + lm, n1 = n0 + 16;
    float bi0 = bias[n0], bi1 = bias[n1];
    int mrow = mtile + hi * 8;
#pragma unroll
    for (int r = 0; r < 8; ++r) {
        int m0 = mrow + r, m1 = m0 + 16;
        int bb0 = m0 >> 12, p0 = m0 & 4095;
        int bb1 = m1 >> 12, p1 = m1 & 4095;
        out[((size_t)(bb0 * Cc + n0)) * HW + p0] = c00[r] + bi0;
        out[((size_t)(bb0 * Cc + n1)) * HW + p0] = c01[r] + bi1;
        out[((size_t)(bb1 * Cc + n0)) * HW + p1] = c10[r] + bi0;
        out[((size_t)(bb1 * Cc + n1)) * HW + p1] = c11[r] + bi1;
    }
}

// ---------------------------------------------------------------------------
// Per-batch mean / sum-of-squares partials (layer-norm over C,H,W).
// Each block reduces 1024 contiguous elements (never crosses a batch:
// 2^20 elems per batch) and atomically accumulates into stats[b*2 + {0,1}].
// ---------------------------------------------------------------------------
__global__ void __launch_bounds__(256)
reduce_stats_kernel(const float* __restrict__ in, float* __restrict__ stats) {
    __shared__ float ss[256];
    __shared__ float sq[256];
    size_t base = (size_t)blockIdx.x * 1024;
    int b = (int)(base >> 20);
    float s = 0.0f, q = 0.0f;
#pragma unroll
    for (int i = 0; i < 4; ++i) {
        float v = in[base + threadIdx.x + i * 256];
        s += v; q += v * v;
    }
    ss[threadIdx.x] = s; sq[threadIdx.x] = q;
    __syncthreads();
    for (int st = 128; st > 0; st >>= 1) {
        if ((int)threadIdx.x < st) {
            ss[threadIdx.x] += ss[threadIdx.x + st];
            sq[threadIdx.x] += sq[threadIdx.x + st];
        }
        __syncthreads();
    }
    if (threadIdx.x == 0) {
        atomicAdd(&stats[b * 2 + 0], ss[0]);
        atomicAdd(&stats[b * 2 + 1], sq[0]);
    }
}

// Residual add (h + x) -> r, fused with stats accumulation for the final norm.
__global__ void __launch_bounds__(256)
resid_stats_kernel(const float* __restrict__ h, const float* __restrict__ x,
                   float* __restrict__ r, float* __restrict__ stats) {
    __shared__ float ss[256];
    __shared__ float sq[256];
    size_t base = (size_t)blockIdx.x * 1024;
    int b = (int)(base >> 20);
    float s = 0.0f, q = 0.0f;
#pragma unroll
    for (int i = 0; i < 4; ++i) {
        size_t idx = base + threadIdx.x + i * 256;
        float v = h[idx] + x[idx];
        r[idx] = v;
        s += v; q += v * v;
    }
    ss[threadIdx.x] = s; sq[threadIdx.x] = q;
    __syncthreads();
    for (int st = 128; st > 0; st >>= 1) {
        if ((int)threadIdx.x < st) {
            ss[threadIdx.x] += ss[threadIdx.x + st];
            sq[threadIdx.x] += sq[threadIdx.x + st];
        }
        __syncthreads();
    }
    if (threadIdx.x == 0) {
        atomicAdd(&stats[b * 2 + 0], ss[0]);
        atomicAdd(&stats[b * 2 + 1], sq[0]);
    }
}

// Apply norm (mean/var from stats) + optional ReLU.
__global__ void __launch_bounds__(256)
norm_act_kernel(const float* __restrict__ in, const float* __restrict__ gamma,
                const float* __restrict__ beta, const float* __restrict__ stats,
                float* __restrict__ out, int relu) {
    size_t i = (size_t)blockIdx.x * 256 + threadIdx.x;
    int b = (int)(i >> 20);
    int c = (int)((i >> 12) & 255);
    const float n = (float)(Cc * HW);
    float mean = stats[b * 2 + 0] / n;
    float var  = stats[b * 2 + 1] / n - mean * mean;
    float rs   = rsqrtf(var + EPSf);
    float v = (in[i] - mean) * rs * gamma[c] + beta[c];
    if (relu) v = fmaxf(v, 0.0f);
    out[i] = v;
}

// ---------------------------------------------------------------------------
extern "C" void kernel_launch(void* const* d_in, const int* in_sizes, int n_in,
                              void* d_out, int out_size, void* d_ws, size_t ws_size,
                              hipStream_t stream) {
    const float* x   = (const float*)d_in[0];
    const float* ow1 = (const float*)d_in[1];
    const float* ob1 = (const float*)d_in[2];
    const float* dw1 = (const float*)d_in[3];
    const float* db1 = (const float*)d_in[4];
    const float* g1  = (const float*)d_in[5];
    const float* b1  = (const float*)d_in[6];
    const float* ow2 = (const float*)d_in[7];
    const float* ob2 = (const float*)d_in[8];
    const float* dw2 = (const float*)d_in[9];
    const float* db2 = (const float*)d_in[10];
    const float* g2  = (const float*)d_in[11];
    const float* b2  = (const float*)d_in[12];
    const float* g3  = (const float*)d_in[13];
    const float* b3  = (const float*)d_in[14];
    float* out = (float*)d_out;

    char* ws = (char*)d_ws;
    _Float16* A    = (_Float16*)(ws + 0ull);            // 75,497,472 B
    _Float16* Bt1  = (_Float16*)(ws + 75497472ull);     //  1,179,648 B
    _Float16* Bt2  = (_Float16*)(ws + 76677120ull);     //  1,179,648 B
    float*    off  = (float*)   (ws + 77856768ull);     //  1,179,648 B
    float*    P0   = (float*)   (ws + 79036416ull);     // 16,777,216 B
    float*    P1   = (float*)   (ws + 95813632ull);     // 16,777,216 B
    float*    st   = (float*)   (ws + 112590848ull);    //        96 B

    const int ELEMS_BLOCKS = (Bn * Cc * HW) / 256;      // 16384 (1 elem/thread)
    const int RED_BLOCKS   = (Bn * Cc * HW) / 1024;     // 4096 (1024 elems/block)

    zero_stats_kernel<<<1, 32, 0, stream>>>(st);
    pack_w_kernel<<<(Cc * KD + 255) / 256, 256, 0, stream>>>(dw1, dw2, Bt1, Bt2);

    // ---- layer 1 ----
    off_conv_kernel<<<NPIX / 256, 256, 0, stream>>>(x, ow1, ob1, off);
    sample_pack_kernel<<<(NPIX * K2c) / 256, 256, 0, stream>>>(x, off, A);
    gemm_wmma_kernel<<<NPIX / 32, 256, 0, stream>>>(A, Bt1, db1, P0);
    reduce_stats_kernel<<<RED_BLOCKS, 256, 0, stream>>>(P0, st + 0);
    norm_act_kernel<<<ELEMS_BLOCKS, 256, 0, stream>>>(P0, g1, b1, st + 0, P1, 1);

    // ---- layer 2 ----
    off_conv_kernel<<<NPIX / 256, 256, 0, stream>>>(P1, ow2, ob2, off);
    sample_pack_kernel<<<(NPIX * K2c) / 256, 256, 0, stream>>>(P1, off, A);
    gemm_wmma_kernel<<<NPIX / 32, 256, 0, stream>>>(A, Bt2, db2, P0);
    reduce_stats_kernel<<<RED_BLOCKS, 256, 0, stream>>>(P0, st + 8);
    norm_act_kernel<<<ELEMS_BLOCKS, 256, 0, stream>>>(P0, g2, b2, st + 8, P0, 1);

    // ---- residual + final norm ----
    resid_stats_kernel<<<RED_BLOCKS, 256, 0, stream>>>(P0, x, P1, st + 16);
    norm_act_kernel<<<ELEMS_BLOCKS, 256, 0, stream>>>(P1, g3, b3, st + 16, out, 0);
}
